// ResNet18_38594576122520
// MI455X (gfx1250) — matrix operand order repaired and checked by reading, not compiled, not run on hardware
//
#include <hip/hip_runtime.h>
#include <hip/hip_bf16.h>

typedef __attribute__((ext_vector_type(16))) _Float16 v16h;
typedef __attribute__((ext_vector_type(8)))  _Float16 v8h;
typedef __attribute__((ext_vector_type(8)))  float    v8f;

#define NB 1024           // batch
#define SP 64             // 8*8 spatial at the WMMA stages

// ---------------------------------------------------------------------------
// Fused conv(KxK, stride1) + BN + ReLU + 2x2 maxpool (+ReLU, idempotent)
// ---------------------------------------------------------------------------
__global__ void conv_pool_k(const float* __restrict__ x, const float* __restrict__ w,
                            const float* __restrict__ g, const float* __restrict__ be,
                            const float* __restrict__ mn, const float* __restrict__ vr,
                            float* __restrict__ out,
                            int Cin, int Cout, int K, int pad, int Hin, int Hout) {
  int idx = blockIdx.x * blockDim.x + threadIdx.x;
  int total = NB * Cout * Hout * Hout;
  if (idx >= total) return;
  int ox = idx % Hout; int t = idx / Hout;
  int oy = t % Hout;   t /= Hout;
  int co = t % Cout;   int b = t / Cout;
  float sc = g[co] * rsqrtf(vr[co] + 1e-5f);
  float sh = be[co] - mn[co] * sc;
  float best = 0.f;                       // relu floor
  for (int dy = 0; dy < 2; ++dy)
    for (int dx = 0; dx < 2; ++dx) {
      int cy = 2 * oy + dy, cx = 2 * ox + dx;
      float acc = 0.f;
      for (int ci = 0; ci < Cin; ++ci)
        for (int r = 0; r < K; ++r) {
          int iy = cy + r - pad; if ((unsigned)iy >= (unsigned)Hin) continue;
          for (int s = 0; s < K; ++s) {
            int ix = cx + s - pad; if ((unsigned)ix >= (unsigned)Hin) continue;
            acc += x[((b * Cin + ci) * Hin + iy) * Hin + ix] *
                   w[((co * Cin + ci) * K + r) * K + s];
          }
        }
      float yv = acc * sc + sh;
      if (yv > best) best = yv;
    }
  out[idx] = best;
}

// ---------------------------------------------------------------------------
// Generic direct conv + optional bias + optional BN + activation
// act: 0=none 1=relu 2=sigmoid
// ---------------------------------------------------------------------------
__global__ void conv_direct_k(const float* __restrict__ x, const float* __restrict__ w,
                              const float* __restrict__ bias, const float* __restrict__ g,
                              const float* __restrict__ be, const float* __restrict__ mn,
                              const float* __restrict__ vr, float* __restrict__ out,
                              int Cin, int Cout, int K, int stride, int pad,
                              int Hin, int Hout, int act) {
  int idx = blockIdx.x * blockDim.x + threadIdx.x;
  int total = NB * Cout * Hout * Hout;
  if (idx >= total) return;
  int ox = idx % Hout; int t = idx / Hout;
  int oy = t % Hout;   t /= Hout;
  int co = t % Cout;   int b = t / Cout;
  float acc = 0.f;
  for (int ci = 0; ci < Cin; ++ci)
    for (int r = 0; r < K; ++r) {
      int iy = oy * stride + r - pad; if ((unsigned)iy >= (unsigned)Hin) continue;
      for (int s = 0; s < K; ++s) {
        int ix = ox * stride + s - pad; if ((unsigned)ix >= (unsigned)Hin) continue;
        acc += x[((b * Cin + ci) * Hin + iy) * Hin + ix] *
               w[((co * Cin + ci) * K + r) * K + s];
      }
    }
  float sc = 1.f, sh = 0.f;
  if (g) { sc = g[co] * rsqrtf(vr[co] + 1e-5f); sh = be[co] - mn[co] * sc;
           if (bias) sh += bias[co] * sc; }
  else if (bias) sh = bias[co];
  float v2 = acc * sc + sh;
  if (act == 1) v2 = fmaxf(v2, 0.f);
  else if (act == 2) v2 = 1.f / (1.f + expf(-v2));
  out[idx] = v2;
}

// ---------------------------------------------------------------------------
// Weight prepack: f32 [Cout, Ktot] (flat OIHW == im2col K-order) -> f16
// [CoutPad, Kp] zero padded.  total = CoutPad*Kp.
// ---------------------------------------------------------------------------
__global__ void wpack_k(const float* __restrict__ w, _Float16* __restrict__ wp,
                        int Cout, int Ktot, int Kp, int total) {
  int idx = blockIdx.x * blockDim.x + threadIdx.x;
  if (idx >= total) return;
  int co = idx / Kp, k = idx - co * Kp;
  float v = (co < Cout && k < Ktot) ? w[co * Ktot + k] : 0.f;
  wp[idx] = (_Float16)v;
}

// ---------------------------------------------------------------------------
// Implicit-GEMM conv on v_wmma_f32_16x16x32_f16 with LDS-staged A tile.
// Block = one 16-row M tile x (blockDim/32) N tiles (one wave each).
// A tile (16x32 f16, 1KB) built cooperatively in LDS per K step; fragments
// read back as two ds_load_b128 per lane.  B read as one aligned v16h from
// the prepacked f16 weights.  Epilogue fuses bias+BN, residual add, ReLU.
// ---------------------------------------------------------------------------
__global__ void conv_wmma2_k(const float* __restrict__ x, const _Float16* __restrict__ wp,
                             const float* __restrict__ bias, const float* __restrict__ g,
                             const float* __restrict__ be, const float* __restrict__ mn,
                             const float* __restrict__ vr, const float* __restrict__ res,
                             float* __restrict__ out,
                             int Cin, int Cout, int kw3, int relu, int Kp) {
  __shared__ alignas(64) _Float16 As[512];
  int mBase = blockIdx.x * 16;
  int waveId = threadIdx.x >> 5, lane = threadIdx.x & 31;
  int half = lane >> 4, lr = lane & 15;
  int co = waveId * 16 + lr;                 // B column owned by this lane
  int Ktot = kw3 ? Cin * 9 : Cin;
  v8f acc = {};
  for (int kk = 0; kk < Ktot; kk += 32) {
    // cooperative im2col A-tile build (shared by all waves of the block)
    for (int e = threadIdx.x; e < 512; e += blockDim.x) {
      int r = e >> 5, kl = e & 31;
      int k = kk + kl;
      float val = 0.f;
      if (k < Ktot) {
        int m = mBase + r, b = m >> 6, p = m & 63, y = p >> 3, xx = p & 7;
        int ci, iy, ix;
        if (kw3) { ci = k / 9; int rs = k - ci * 9;
                   iy = y + rs / 3 - 1; ix = xx + rs % 3 - 1; }
        else     { ci = k; iy = y; ix = xx; }
        if ((unsigned)iy < 8u && (unsigned)ix < 8u)
          val = x[((b * Cin + ci) << 6) + (iy << 3) + ix];
      }
      As[e] = (_Float16)val;
    }
    __syncthreads();
    // A fragment: lane half h -> K {16h..16h+15} contiguous, rows = lr
    v16h a = *(const v16h*)&As[lr * 32 + half * 16];
    // B fragment: 16 contiguous K halves of column co (padded rows are zero)
    v16h bfr = *(const v16h*)&wp[co * Kp + kk + half * 16];
    acc = __builtin_amdgcn_wmma_f32_16x16x32_f16(false, a, false, bfr,
                                                 (short)0, acc, false, false);
    __syncthreads();
  }
  float sc = 1.f, sh = 0.f;
  if (co < Cout) {
    if (g) { sc = g[co] * rsqrtf(vr[co] + 1e-5f); sh = be[co] - mn[co] * sc;
             if (bias) sh += bias[co] * sc; }
    else if (bias) sh = bias[co];
  }
#pragma unroll
  for (int r = 0; r < 8; ++r) {              // D: vgpr r -> M = r + half*8
    int m = mBase + r + half * 8;
    int b = m >> 6, p = m & 63;
    if (co < Cout) {
      float vo = acc[r] * sc + sh;
      int oi = ((b * Cout + co) << 6) + p;
      if (res) vo += res[oi];
      if (relu) vo = fmaxf(vo, 0.f);
      out[oi] = vo;
    }
  }
}

// ---------------------------------------------------------------------------
// FC GEMM on WMMA with LDS-staged A: out[1024,N] = X[1024,K] W[N,K]^T + bias
// grid = (64 M-tiles, nTile groups); blockDim/32 waves each own one N tile.
// ---------------------------------------------------------------------------
__global__ void gemm_wmma2_k(const float* __restrict__ X, const _Float16* __restrict__ wp,
                             const float* __restrict__ bias, float* __restrict__ out,
                             int K, int N, int relu, int Kp) {
  __shared__ alignas(64) _Float16 As[512];
  int mBase = blockIdx.x * 16;
  int waveId = threadIdx.x >> 5, lane = threadIdx.x & 31;
  int nTile = blockIdx.y * (blockDim.x >> 5) + waveId;
  int half = lane >> 4, lr = lane & 15;
  int n = nTile * 16 + lr;
  v8f acc = {};
  for (int kk = 0; kk < K; kk += 32) {        // K is a multiple of 32 here
    for (int e = threadIdx.x; e < 512; e += blockDim.x) {
      int r = e >> 5, kl = e & 31;
      As[e] = (_Float16)X[(mBase + r) * K + kk + kl];
    }
    __syncthreads();
    v16h a = *(const v16h*)&As[lr * 32 + half * 16];
    v16h bfr = *(const v16h*)&wp[n * Kp + kk + half * 16];
    acc = __builtin_amdgcn_wmma_f32_16x16x32_f16(false, a, false, bfr,
                                                 (short)0, acc, false, false);
    __syncthreads();
  }
#pragma unroll
  for (int r = 0; r < 8; ++r) {
    int m = mBase + r + half * 8;
    if (n < N) {
      float vv = acc[r] + (bias ? bias[n] : 0.f);
      if (relu) vv = fmaxf(vv, 0.f);
      out[m * N + n] = vv;
    }
  }
}

// ---------------------------------------------------------------------------
// A-block / B-block elementwise helpers
// ---------------------------------------------------------------------------
__global__ void channel_mean_k(const float* __restrict__ x, float* __restrict__ out, int C) {
  int idx = blockIdx.x * blockDim.x + threadIdx.x;
  if (idx >= NB * SP) return;
  int p = idx & 63, b = idx >> 6;
  float s = 0.f;
  for (int c = 0; c < C; ++c) s += x[((b * C + c) << 6) + p];
  out[idx] = s / (float)C;
}

// S[b,c,y,x] = F0[B-1-b, c%Ch, y, x] * a0[B-1-b, 0, y, x];  S has 2*Ch channels
__global__ void ablock_gather_k(const float* __restrict__ F0, const float* __restrict__ a0,
                                float* __restrict__ S, int Ch) {
  int idx = blockIdx.x * blockDim.x + threadIdx.x;
  int total = NB * 2 * Ch * SP;
  if (idx >= total) return;
  int p = idx & 63; int t = idx >> 6;
  int c = t % (2 * Ch); int b = t / (2 * Ch);
  int rb = (NB - 1) - b;
  S[idx] = F0[((rb * Ch + (c % Ch)) << 6) + p] * a0[rb * SP + p];
}

// B-block coefficients: one thread per batch element (C fixed = 32)
__global__ void bblock_coef_k(const float* __restrict__ x, const float* __restrict__ f1w,
                              const float* __restrict__ g, const float* __restrict__ be,
                              const float* __restrict__ mn, const float* __restrict__ vr,
                              const float* __restrict__ f2w,
                              float* __restrict__ coefA, float* __restrict__ coefB) {
  int b = blockIdx.x * blockDim.x + threadIdx.x;
  if (b >= NB) return;
  float Vf[32];
  for (int c = 0; c < 32; ++c) {
    const float* p = x + ((b * 32 + c) << 6);
    float s = 0.f;
    for (int i = 0; i < 64; ++i) s += p[i];
    Vf[c] = s * (1.f / 64.f);
  }
  float z[8];
  for (int j = 0; j < 8; ++j) {
    float s = 0.f;
    for (int c = 0; c < 32; ++c) s += f1w[j * 32 + c] * Vf[c];
    float sc = g[j] * rsqrtf(vr[j] + 1e-5f);
    z[j] = fmaxf(s * sc + be[j] - mn[j] * sc, 0.f);
  }
  float t2[32];
  for (int c = 0; c < 32; ++c) {
    float s = 0.f;
    for (int j = 0; j < 8; ++j) s += f2w[c * 8 + j] * z[j];
    t2[c] = s;
  }
  for (int cc = 0; cc < 16; ++cc) {
    float u = Vf[cc], w2 = Vf[16 + cc];
    float mx = fmaxf(u, w2);
    float e0 = expf(u - mx), e1 = expf(w2 - mx);
    float a0 = e0 / (e0 + e1);
    float p0 = t2[cc], p1 = t2[16 + cc];
    float mb = fmaxf(p0, p1);
    float f0 = expf(p0 - mb), f1v = expf(p1 - mb);
    float s = f0 + f1v;
    coefA[b * 16 + cc] = a0 + f0 / s;
    coefB[b * 16 + cc] = f1v / s;
  }
}

// out[:, :16]  = relu(x1 * coefA);  out[:, 16:] = relu(V1[1] + x2 * coefB)
__global__ void bblock_apply_k(const float* __restrict__ x, const float* __restrict__ coefA,
                               const float* __restrict__ coefB, float* __restrict__ out) {
  int idx = blockIdx.x * blockDim.x + threadIdx.x;
  if (idx >= NB * 16 * SP) return;
  int p = idx & 63; int t = idx >> 6;
  int cc = t & 15; int b = t >> 4;
  float x1 = x[((b * 32 + cc) << 6) + p];
  out[((b * 32 + cc) << 6) + p] = fmaxf(x1 * coefA[b * 16 + cc], 0.f);
  float v11 = x[((32 + cc) << 6) + p] * coefA[16 + cc];      // V1 at batch index 1
  float x2 = x[((b * 32 + 16 + cc) << 6) + p];
  out[((b * 32 + 16 + cc) << 6) + p] = fmaxf(v11 + x2 * coefB[b * 16 + cc], 0.f);
}

__global__ void concat_k(const float* __restrict__ a, const float* __restrict__ b,
                         float* __restrict__ out, int Ca, int Cb) {
  int idx = blockIdx.x * blockDim.x + threadIdx.x;
  int total = NB * (Ca + Cb) * SP;
  if (idx >= total) return;
  int p = idx & 63; int t = idx >> 6;
  int c = t % (Ca + Cb); int bb = t / (Ca + Cb);
  out[idx] = (c < Ca) ? a[((bb * Ca + c) << 6) + p]
                      : b[((bb * Cb + (c - Ca)) << 6) + p];
}

// ---------------------------------------------------------------------------
// Host orchestration
// ---------------------------------------------------------------------------
extern "C" void kernel_launch(void* const* d_in, const int* in_sizes, int n_in,
                              void* d_out, int out_size, void* d_ws, size_t ws_size,
                              hipStream_t stream) {
  (void)in_sizes; (void)n_in; (void)out_size; (void)ws_size;
  auto IN = [&](int i) { return (const float*)d_in[i]; };

  // ---- workspace arena (floats, 64B-aligned blocks) ----
  float* wsf = (float*)d_ws;
  size_t off = 0;
  auto alloc = [&](size_t n) {
    off = (off + 15) & ~(size_t)15;
    float* p = wsf + off; off += n; return p;
  };
  const size_t U = (size_t)NB * SP;         // 65536 per channel-plane
  float* topA  = alloc(U * 32);
  float* cenA  = alloc(U * 32);
  float* cenB  = alloc(U * 32);
  float* botB  = alloc(U * 32);
  float* p0    = alloc((size_t)NB * 4 * 32 * 32);
  float* p1    = alloc((size_t)NB * 8 * 16 * 16);
  float* p2    = alloc(U * 16);
  float* t24   = alloc(U * 24);
  float* sc24  = alloc(U * 24);
  float* o24   = alloc(U * 24);
  float* t32   = alloc(U * 32);
  float* sc32  = alloc(U * 32);
  float* o32   = alloc(U * 32);
  float* aF0   = alloc(U * 32);
  float* aM    = alloc(U);
  float* aT    = alloc(U * 16);
  float* aSig  = alloc(U);
  float* aS    = alloc(U * 64);
  float* cat64 = alloc(U * 64);
  float* ab64  = alloc(U * 64);
  float* t64   = alloc(U * 64);
  float* topR  = alloc(U * 64);
  float* botR  = alloc(U * 64);
  float* cat128= alloc(U * 128);
  float* t128  = alloc(U * 128);
  float* o128  = alloc(U * 128);
  float* fc1o  = alloc((size_t)NB * 512);
  float* fc2o  = alloc((size_t)NB * 128);
  float* coefA = alloc((size_t)NB * 16);
  float* coefB = alloc((size_t)NB * 16);
  _Float16* wpack = (_Float16*)alloc(2097152);   // 8.4MB: fits fc1 (512x8192 f16)

  auto blocks_for = [](long total) { return (int)((total + 255) / 256); };

  auto conv_pool = [&](const float* x, int wi, float* out, int Cin, int Cout,
                       int K, int pad, int Hin, int Hout) {
    long total = (long)NB * Cout * Hout * Hout;
    conv_pool_k<<<blocks_for(total), 256, 0, stream>>>(
        x, IN(wi), IN(wi + 1), IN(wi + 2), IN(wi + 3), IN(wi + 4), out,
        Cin, Cout, K, pad, Hin, Hout);
  };
  auto conv_direct = [&](const float* x, const float* w, const float* bias,
                         const float* g, const float* be, const float* mn, const float* vr,
                         float* out, int Cin, int Cout, int K, int stride, int pad,
                         int Hin, int Hout, int act) {
    long total = (long)NB * Cout * Hout * Hout;
    conv_direct_k<<<blocks_for(total), 256, 0, stream>>>(
        x, w, bias, g, be, mn, vr, out, Cin, Cout, K, stride, pad, Hin, Hout, act);
  };
  auto conv_wmma = [&](const float* x, const float* w, const float* bias,
                       const float* g, const float* be, const float* mn, const float* vr,
                       const float* res, float* out, int Cin, int Cout, int kw3, int relu) {
    int Ktot = kw3 ? Cin * 9 : Cin;
    int Kp = (Ktot + 31) & ~31;
    int nT = (Cout + 15) / 16;                 // <= 8 for all conv layers here
    int total = nT * 16 * Kp;
    wpack_k<<<blocks_for(total), 256, 0, stream>>>(w, wpack, Cout, Ktot, Kp, total);
    conv_wmma2_k<<<4096, nT * 32, 0, stream>>>(
        x, wpack, bias, g, be, mn, vr, res, out, Cin, Cout, kw3, relu, Kp);
  };
  auto gemm = [&](const float* X, const float* W, const float* bias, float* out,
                  int K, int N, int relu) {
    int nTiles = (N + 15) / 16;
    int total = nTiles * 16 * K;               // K already a multiple of 32
    wpack_k<<<blocks_for(total), 256, 0, stream>>>(W, wpack, N, K, K, total);
    int nW = nTiles < 8 ? nTiles : 8;
    dim3 grid(64, (nTiles + nW - 1) / nW);
    gemm_wmma2_k<<<grid, nW * 32, 0, stream>>>(X, wpack, bias, out, K, N, relu, K);
  };
  // residual block: relu(bn1(conv1)) -> bn2(conv2) ; shortcut ; add+relu fused
  auto resblk = [&](const float* in, int Cin, int Cout, int base, bool hasE,
                    float* t, float* sc, float* out) {
    conv_wmma(in, IN(base), IN(base + 1), IN(base + 2), IN(base + 3), IN(base + 4),
              IN(base + 5), nullptr, t, Cin, Cout, 1, 1);
    const float* scp = in;
    if (hasE) {
      conv_wmma(in, IN(base + 12), IN(base + 13), IN(base + 14), IN(base + 15),
                IN(base + 16), IN(base + 17), nullptr, sc, Cin, Cout, 0, 0);
      scp = sc;
    }
    conv_wmma(t, IN(base + 6), IN(base + 7), IN(base + 8), IN(base + 9), IN(base + 10),
              IN(base + 11), scp, out, Cout, Cout, 1, 1);
  };
  // A-block (dead-code-eliminated reference: only F0 path, batch flip, gate)
  auto ablock = [&](const float* in, int pl, int base, float* out) {
    int Ch = pl / 2, q = pl / 4;
    conv_wmma(in, IN(base), IN(base + 1), nullptr, nullptr, nullptr, nullptr,
              nullptr, aF0, pl, Ch, 0, 0);                       // c1 1x1
    channel_mean_k<<<blocks_for(NB * SP), 256, 0, stream>>>(aF0, aM, Ch);
    conv_direct(aM, IN(base + 2), IN(base + 3), nullptr, nullptr, nullptr, nullptr,
                aT, 1, q, 3, 1, 1, 8, 8, 1);                     // c5 relu
    conv_direct(aT, IN(base + 4), IN(base + 5), nullptr, nullptr, nullptr, nullptr,
                aSig, q, 1, 3, 1, 1, 8, 8, 2);                   // c6 sigmoid
    ablock_gather_k<<<blocks_for((long)NB * pl * SP), 256, 0, stream>>>(aF0, aSig, aS, Ch);
    conv_wmma(aS, IN(base + 6), IN(base + 7), nullptr, nullptr, nullptr, nullptr,
              nullptr, out, pl, pl, 1, 1);                       // c7 relu
  };
  auto bblock = [&](const float* in, float* out) {
    bblock_coef_k<<<4, 256, 0, stream>>>(in, IN(194), IN(195), IN(196), IN(197),
                                         IN(198), IN(199), coefA, coefB);
    bblock_apply_k<<<blocks_for((long)NB * 16 * SP), 256, 0, stream>>>(in, coefA, coefB, out);
  };
  auto concat = [&](const float* a, const float* b, float* out, int Ca, int Cb) {
    concat_k<<<blocks_for((long)NB * (Ca + Cb) * SP), 256, 0, stream>>>(a, b, out, Ca, Cb);
  };
  // 64x64 stem -> [B,32,8,8]
  auto stem64 = [&](const float* in, int rp1, int rp2, int rp4, int blkA, int blkB) {
    conv_pool(in, rp1, p0, 1, 4, 5, 2, 64, 32);
    conv_pool(p0, rp2, p1, 4, 8, 3, 1, 32, 16);
    conv_direct(p1, IN(rp4), IN(rp4 + 1), IN(rp4 + 2), IN(rp4 + 3), IN(rp4 + 4),
                IN(rp4 + 5), p2, 8, 16, 3, 2, 1, 16, 8, 0);
    resblk(p2, 16, 24, blkA, true, t24, sc24, o24);
    resblk(o24, 24, 32, blkB, true, t32, sc32, o32);
  };

  const float* ms     = IN(0);
  const float* pan    = IN(1);
  const float* mshpan = IN(2);

  // ---- top branch (pan) ----
  stem64(pan, 3, 8, 28, 46, 64);
  ablock(o32, 32, 178, topA);
  // ---- center branch (mshpan) ----
  stem64(mshpan, 13, 18, 34, 82, 100);
  ablock(o32, 32, 178, cenA);
  bblock(o32, cenB);
  // ---- bottom branch (ms) ----
  conv_direct(ms, IN(23), nullptr, IN(24), IN(25), IN(26), IN(27),
              p1, 4, 8, 3, 1, 1, 16, 16, 1);                     // R3
  conv_direct(p1, IN(40), IN(41), IN(42), IN(43), IN(44), IN(45),
              p2, 8, 16, 3, 2, 1, 16, 8, 0);                     // RP3_4
  resblk(p2, 16, 24, 118, true, t24, sc24, o24);                 // blk3_1
  resblk(o24, 24, 32, 136, true, t32, sc32, o32);                // blk3_2
  bblock(o32, botB);

  // ---- fusion ----
  concat(topA, cenA, cat64, 32, 32);
  ablock(cat64, 64, 186, ab64);
  resblk(ab64, 64, 64, 154, false, t64, nullptr, topR);          // blk3
  concat(cenB, botB, cat64, 32, 32);
  ablock(cat64, 64, 186, ab64);
  resblk(ab64, 64, 64, 154, false, t64, nullptr, botR);          // blk3 (shared)
  concat(topR, botR, cat128, 64, 64);
  resblk(cat128, 128, 128, 166, false, t128, nullptr, o128);     // blk4

  // ---- FC head (NCHW flatten == reshape(B,-1)) ----
  gemm(o128, IN(200), IN(201), fc1o, 8192, 512, 1);
  gemm(fc1o, IN(202), IN(203), fc2o, 512, 128, 1);
  gemm(fc2o, IN(204), IN(205), (float*)d_out, 128, 12, 0);
}
